// PoolHiddenNet_74577812127758
// MI455X (gfx1250) — matrix operand order repaired and checked by reading, not compiled, not run on hardware
//
#include <hip/hip_runtime.h>
#include <hip/hip_bf16.h>

// ---------------------------------------------------------------------------
// PoolHiddenNet on MI455X (gfx1250), wave32 + v_wmma_f32_16x16x32_f16.
//
// G=128 P=32 H=64 E=64 D1=512 D2=1024, N=4096.
//
// Math (biases cancel in train-mode BN):
//   q[n]    = pos[n] @ Ws                       (64)
//   u[n]    = q@W1a + hid@W1b + (spd*Wv)@W1c    (512)  "j-side"
//   t[n]    = q@W1a                             (512)  "i-side"
//   BN1:  mean = mean_j u - mean_i t,  var = var(u)+var(t)   (exact)
//   h1[g,i,j] = relu(A[g,j] + B[g,i]),  A=(u-ū)s+be1, B=-(t-t̄)s, s=g1*rsqrt(var+eps)
//   D = h1 @ W2  (f16 WMMA, f32 acc);  BN2 stats over 1024 rows/group (pass 1)
//   out[g,i,c] = max_j relu(D*scale + shift)    (pass 2, fused epilogue)
// ---------------------------------------------------------------------------

typedef _Float16 half16 __attribute__((ext_vector_type(16)));
typedef _Float16 half2t __attribute__((ext_vector_type(2)));
typedef float    f32x8  __attribute__((ext_vector_type(8)));

#define GRP 128
#define PED 32
#define D1  512
#define D2  1024

// ---------------- prep: W2 (f32 [512,1024]) -> W2t (f16 [1024,512]) ---------
__global__ void k_w2t(const float* __restrict__ W2, _Float16* __restrict__ W2t) {
  int n = blockIdx.x; // 0..1023
  for (int k = threadIdx.x; k < D1; k += blockDim.x)
    W2t[(size_t)n * D1 + k] = (_Float16)W2[(size_t)k * D2 + n];
}

// ---------------- prep: per-ped u,t vectors ---------------------------------
__global__ void k_ut(const float* __restrict__ hid, const float* __restrict__ pos,
                     const float* __restrict__ spd, const float* __restrict__ Ws,
                     const float* __restrict__ Wv, const float* __restrict__ W1,
                     float* __restrict__ u, float* __restrict__ t) {
  int n = blockIdx.y;        // ped 0..4095
  int halfb = blockIdx.x;    // 0..1 (which 256 of 512 channels)
  int tid = threadIdx.x;
  __shared__ float z[192];   // [q(64) | hid(64) | spd*Wv(64)]
  if (tid < 64) {
    float p0 = pos[n * 2 + 0], p1 = pos[n * 2 + 1];
    z[tid] = p0 * Ws[tid] + p1 * Ws[64 + tid];
  } else if (tid < 128) {
    z[tid] = hid[(size_t)n * 64 + (tid - 64)];
  } else if (tid < 192) {
    z[tid] = spd[n] * Wv[tid - 128];
  }
  __syncthreads();
  int c = halfb * 256 + tid; // 0..511
  float acc = 0.f;
  for (int k = 0; k < 64; k++)  acc += z[k] * W1[(size_t)k * D1 + c];
  t[(size_t)n * D1 + c] = acc;
  for (int k = 64; k < 192; k++) acc += z[k] * W1[(size_t)k * D1 + c];
  u[(size_t)n * D1 + c] = acc;
}

// ---------------- prep: BN1 stats (closed form) -> A,B f16 ------------------
__global__ void k_ab(const float* __restrict__ u, const float* __restrict__ t,
                     const float* __restrict__ g1, const float* __restrict__ be1,
                     _Float16* __restrict__ Ah, _Float16* __restrict__ Bh) {
  int g = blockIdx.x, c = threadIdx.x; // 512 threads
  float su = 0, squ = 0, st = 0, sqt = 0;
  for (int j = 0; j < PED; j++) {
    float v = u[((size_t)g * PED + j) * D1 + c]; su += v; squ += v * v;
    float w = t[((size_t)g * PED + j) * D1 + c]; st += w; sqt += w * w;
  }
  float mu = su * (1.f / PED), mt = st * (1.f / PED);
  float var = (squ * (1.f / PED) - mu * mu) + (sqt * (1.f / PED) - mt * mt);
  float s = rsqrtf(fmaxf(var, 0.f) + 1e-5f) * g1[c];
  float bb = be1[c];
  for (int j = 0; j < PED; j++) {
    size_t o = ((size_t)g * PED + j) * D1 + c;
    Ah[o] = (_Float16)((u[o] - mu) * s + bb);
    Bh[o] = (_Float16)(-(t[o] - mt) * s);
  }
}

// ---------------- BN2 stats -> per (g,channel) scale/shift ------------------
__global__ void k_scale(const float* __restrict__ ssum, const float* __restrict__ ssq,
                        const float* __restrict__ g2, const float* __restrict__ be2,
                        float* __restrict__ scl, float* __restrict__ shf) {
  int g = blockIdx.y;
  int c = blockIdx.x * 256 + threadIdx.x;
  float mean = ssum[(size_t)g * D2 + c] * (1.f / 1024.f);
  float var = ssq[(size_t)g * D2 + c] * (1.f / 1024.f) - mean * mean;
  float s = g2[c] * rsqrtf(fmaxf(var, 0.f) + 1e-5f);
  scl[(size_t)g * D2 + c] = s;
  shf[(size_t)g * D2 + c] = be2[c] - mean * s;
}

// ---------------- main GEMM: D = relu(A_j + B_i) @ W2t^T --------------------
// Grid (nt=8, mt=8, g=128), 256 threads = 8 waves (4M x 2N), 128x128 tile.
// PASS 1: accumulate per-channel sum/sumsq of D.  PASS 2: BN+relu+max_j -> out.
template <int PASS>
__global__ __launch_bounds__(256) void gemm_pass(
    const _Float16* __restrict__ Ah, const _Float16* __restrict__ Bh,
    const _Float16* __restrict__ W2t,
    float* __restrict__ ssum, float* __restrict__ ssq,
    const float* __restrict__ scl, const float* __restrict__ shf,
    float* __restrict__ out) {
  const int nt = blockIdx.x, mt = blockIdx.y, g = blockIdx.z;
  const int tid = threadIdx.x;
  const int w = tid >> 5, lane = tid & 31, lm = lane & 15, hi = lane >> 4;
  const int wm = w >> 1, wn = w & 1;

  // padded rows: 520 halves (1040B, 16B aligned) / 40 halves (80B)
  __shared__ __align__(16) _Float16 sA[32 * 520]; // A vectors, all 32 j
  __shared__ __align__(16) _Float16 sB[4 * 520];  // B vectors, i0..i0+3
  __shared__ __align__(16) _Float16 sW[128 * 40]; // W2t panel: 128 n x 32 k
  __shared__ float sStat[256];

  if (PASS == 1) sStat[tid] = 0.f;

  { // stage A (32x512 f16) and B (4x512 f16)
    const uint4* srcA = (const uint4*)(Ah + (size_t)g * PED * D1);
    for (int q = tid; q < 32 * 64; q += 256) {
      int r = q >> 6, s = q & 63;
      *(uint4*)(sA + r * 520 + s * 8) = srcA[r * 64 + s];
    }
    const uint4* srcB = (const uint4*)(Bh + ((size_t)g * PED + mt * 4) * D1);
    for (int q = tid; q < 4 * 64; q += 256) {
      int r = q >> 6, s = q & 63;
      *(uint4*)(sB + r * 520 + s * 8) = srcB[r * 64 + s];
    }
  }

  f32x8 acc[2][4] = {};

  for (int k0 = 0; k0 < D1; k0 += 32) {
    __syncthreads();
    // stage W2t panel [nt*128 .. +128)[k0 .. k0+32)
    for (int q = tid; q < 512; q += 256) {
      int r = q >> 2, s = q & 3;
      *(uint4*)(sW + r * 40 + s * 8) =
          *(const uint4*)(W2t + ((size_t)(nt * 128 + r)) * D1 + k0 + s * 8);
    }
    __syncthreads();

    // B fragments: lane holds N=lm, K = 16*hi + 2v + h
    half16 bf[4];
#pragma unroll
    for (int nf = 0; nf < 4; nf++) {
      int nl = wn * 64 + nf * 16 + lm;
      const unsigned* p = (const unsigned*)(sW + nl * 40 + hi * 16);
      unsigned* d = (unsigned*)&bf[nf];
#pragma unroll
      for (int v = 0; v < 8; v++) d[v] = p[v];
    }

    // A fragments generated on the fly: relu(A[j] + B[i]) in packed f16.
    // Lane holds M=lm (j = mf*16+lm, i = mt*4+wm); K = 16*hi + 2v + h
    half16 af[2];
#pragma unroll
    for (int mf = 0; mf < 2; mf++) {
      int j = mf * 16 + lm;
      const unsigned* pa = (const unsigned*)(sA + j * 520);
      const unsigned* pb = (const unsigned*)(sB + wm * 520);
      unsigned* d = (unsigned*)&af[mf];
#pragma unroll
      for (int v = 0; v < 8; v++) {
        int koff = (hi << 4) + (v << 1);
        int ku = (k0 + koff) >> 1;
        union { unsigned u; half2t h; } xa, xb, r;
        xa.u = pa[ku]; xb.u = pb[ku];
        r.h = xa.h + xb.h;
        r.h.x = r.h.x > (_Float16)0 ? r.h.x : (_Float16)0;
        r.h.y = r.h.y > (_Float16)0 ? r.h.y : (_Float16)0;
        d[v] = r.u;
      }
    }

#pragma unroll
    for (int mf = 0; mf < 2; mf++)
#pragma unroll
      for (int nf = 0; nf < 4; nf++)
        acc[mf][nf] = __builtin_amdgcn_wmma_f32_16x16x32_f16(
            false, af[mf], false, bf[nf], (short)0, acc[mf][nf], false, false);
  }

  if (PASS == 1) {
    __syncthreads();
#pragma unroll
    for (int nf = 0; nf < 4; nf++) {
      float s1 = 0.f, s2 = 0.f;
#pragma unroll
      for (int mf = 0; mf < 2; mf++)
#pragma unroll
        for (int r = 0; r < 8; r++) { float v = acc[mf][nf][r]; s1 += v; s2 += v * v; }
      int nl = wn * 64 + nf * 16 + lm;
      atomicAdd(&sStat[nl], s1);
      atomicAdd(&sStat[128 + nl], s2);
    }
    __syncthreads();
    if (tid < 128) {
      atomicAdd(&ssum[(size_t)g * D2 + nt * 128 + tid], sStat[tid]);
      atomicAdd(&ssq[(size_t)g * D2 + nt * 128 + tid], sStat[128 + tid]);
    }
  } else {
    // Each wave's 32 M-rows are exactly one i (= mt*4+wm), all 32 j.
    int i = mt * 4 + wm;
#pragma unroll
    for (int nf = 0; nf < 4; nf++) {
      int colg = nt * 128 + wn * 64 + nf * 16 + lm;
      float sc = scl[(size_t)g * D2 + colg];
      float sh = shf[(size_t)g * D2 + colg];
      float m = 0.f; // relu => max over j of relu == max(0, raw values)
#pragma unroll
      for (int mf = 0; mf < 2; mf++)
#pragma unroll
        for (int r = 0; r < 8; r++)
          m = fmaxf(m, acc[mf][nf][r] * sc + sh);
      m = fmaxf(m, __shfl_xor(m, 16, 32)); // combine rows 0-7,16-23 with 8-15,24-31
      if (hi == 0) out[((size_t)g * PED + i) * D2 + colg] = m;
    }
  }
}

// ---------------------------------------------------------------------------
extern "C" void kernel_launch(void* const* d_in, const int* in_sizes, int n_in,
                              void* d_out, int out_size, void* d_ws, size_t ws_size,
                              hipStream_t stream) {
  const float* h_states = (const float*)d_in[0];   // [4096,64]
  // d_in[1] seq_start_end: uniform groups, unused
  const float* end_pos = (const float*)d_in[2];    // [4096,2]
  const float* spd     = (const float*)d_in[3];    // [4096,1]
  const float* Ws = (const float*)d_in[4];         // [2,64]
  const float* Wv = (const float*)d_in[6];         // [1,64]
  const float* W1 = (const float*)d_in[8];         // [192,512]
  const float* g1  = (const float*)d_in[10];
  const float* be1 = (const float*)d_in[11];
  const float* W2  = (const float*)d_in[12];       // [512,1024]
  const float* g2  = (const float*)d_in[14];
  const float* be2 = (const float*)d_in[15];
  float* out = (float*)d_out;                      // [4096,1024]

  char* ws = (char*)d_ws;
  const size_t MB = 1u << 20;
  _Float16* W2t = (_Float16*)(ws);                 //  1 MB
  float*    u   = (float*)(ws + 1 * MB);           //  8 MB
  float*    t   = (float*)(ws + 9 * MB);           //  8 MB
  _Float16* Ah  = (_Float16*)(ws + 17 * MB);       //  4 MB
  _Float16* Bh  = (_Float16*)(ws + 21 * MB);       //  4 MB
  float*    ssum = (float*)(ws + 25 * MB);         // 512 KB
  float*    ssq  = (float*)(ws + 25 * MB + 512 * 1024);
  float*    scl  = (float*)(ws + 26 * MB);         // 512 KB
  float*    shf  = (float*)(ws + 26 * MB + 512 * 1024);

  hipMemsetAsync(ws + 25 * MB, 0, 1 * MB, stream); // zero BN2 accumulators

  k_w2t<<<dim3(1024), 128, 0, stream>>>(W2, W2t);
  k_ut<<<dim3(2, 4096), 256, 0, stream>>>(h_states, end_pos, spd, Ws, Wv, W1, u, t);
  k_ab<<<dim3(128), 512, 0, stream>>>(u, t, g1, be1, Ah, Bh);
  gemm_pass<1><<<dim3(8, 8, 128), 256, 0, stream>>>(Ah, Bh, W2t, ssum, ssq, scl, shf, out);
  k_scale<<<dim3(4, 128), 256, 0, stream>>>(ssum, ssq, g2, be2, scl, shf);
  gemm_pass<2><<<dim3(8, 8, 128), 256, 0, stream>>>(Ah, Bh, W2t, ssum, ssq, scl, shf, out);
}
